// MyModel_1700807049891
// MI455X (gfx1250) — matrix operand order repaired and checked by reading, not compiled, not run on hardware
//
#include <hip/hip_runtime.h>
#include <hip/hip_bf16.h>
#include <math.h>

typedef __bf16 bf16;
typedef __attribute__((ext_vector_type(16))) __bf16 v16bf;
typedef __attribute__((ext_vector_type(8)))  __bf16 v8bf;
typedef __attribute__((ext_vector_type(8)))  float  v8f;

#define B_  32
#define T_  256
#define E_  512
#define H_  512
#define G_  2048    /* 4*H */
#define C_  9
#define M_  (B_*T_) /* 8192 rows */
#define NBLK_SCAN 16

__device__ __forceinline__ float sigmf(float x) { return 1.0f / (1.0f + __expf(-x)); }

__device__ __forceinline__ v16bf cat8(v8bf lo, v8bf hi) {
    return __builtin_shufflevector(lo, hi, 0,1,2,3,4,5,6,7,8,9,10,11,12,13,14,15);
}

// async global -> LDS b128 copy (CDNA5 ASYNCcnt path). ldsoff = LDS byte offset
// (low 32 bits of the generic pointer to a __shared__ object, per aperture map).
__device__ __forceinline__ void async_g2l_b128(unsigned ldsoff, const void* gaddr) {
    asm volatile("global_load_async_to_lds_b128 %0, %1, off"
                 :: "v"(ldsoff), "v"((unsigned long long)(uintptr_t)gaddr)
                 : "memory");
}
__device__ __forceinline__ void wait_async0() {
    asm volatile("s_wait_asynccnt 0x0" ::: "memory");
}
__device__ __forceinline__ void wait_async3() {   // <=3 outstanding: previous group landed
    asm volatile("s_wait_asynccnt 0x3" ::: "memory");
}

// ---------------------------------------------------------------- init / prep
__global__ void zero_init(bf16* hF, bf16* hB, unsigned* bar, float* out, int n) {
    int i = blockIdx.x * blockDim.x + threadIdx.x;
    if (i < n) { hF[i] = (bf16)0.f; hB[i] = (bf16)0.f; }
    if (i < 2 * T_) bar[i] = 0u;
    if (i == 0) out[0] = 0.f;
}

__global__ void cvt_bf16(const float* __restrict__ s, bf16* __restrict__ d, int n) {
    int i = blockIdx.x * blockDim.x + threadIdx.x;
    if (i < n) d[i] = (bf16)s[i];
}

__global__ void bias_sum(const float* __restrict__ a, const float* __restrict__ b,
                         float* __restrict__ o, int n) {
    int i = blockIdx.x * blockDim.x + threadIdx.x;
    if (i < n) o[i] = a[i] + b[i];
}

__global__ void embed_gather(const int* __restrict__ ids, const float* __restrict__ tab,
                             bf16* __restrict__ xb) {
    int row = blockIdx.x;
    int id  = ids[row];
    const float* src = &tab[(size_t)id * E_];
    bf16* dst = &xb[(size_t)row * E_];
    for (int e = threadIdx.x; e < E_; e += blockDim.x)
        dst[e] = (id == 0) ? (bf16)0.f : (bf16)src[e];   // emb.at[0].set(0)
}

// ------------------------------- XP = Xbf @ W^T + bias (WMMA bf16, both dirs)
// X: [M][E] bf16 row-major, W: [G][E] bf16 row-major (== B matrix N-major/K-contig)
// 2-stage software pipeline: async-DMA tile k+1 overlaps WMMA on tile k.
__global__ void __launch_bounds__(256)
xproj_gemm(const bf16* __restrict__ X,
           const bf16* __restrict__ Wf, const bf16* __restrict__ Wb,
           const float* __restrict__ biasFp, const float* __restrict__ biasBp,
           float* __restrict__ XPfp, float* __restrict__ XPbp) {
    __shared__ __align__(16) bf16 As[2][128 * 32];
    __shared__ __align__(16) bf16 Bs[2][64 * 32];
    const bf16*  W    = blockIdx.z ? Wb : Wf;
    const float* bias = blockIdx.z ? biasBp : biasFp;
    float*       XP   = blockIdx.z ? XPbp : XPfp;

    const int tid  = threadIdx.x;
    const int lane = tid & 31;
    const int wv   = tid >> 5;
    const int Mbase = blockIdx.y * 128;
    const int Nbase = blockIdx.x * 64;
    const int m0w = (wv & 3) * 32;
    const int n0w = (wv >> 2) * 32;

    v8f acc00 = {}, acc01 = {}, acc10 = {}, acc11 = {};

    const int arow  = m0w + (lane & 15);
    const int koffA = (lane >> 4) * 16;  // A frag: lanes 0-15 K=0..15, 16-31 K=16..31
    const int koffB = (lane >> 4) * 16;  // B frag: lanes 0-15 K=0..15, 16-31 K=16..31
    const int bcol  = n0w + (lane & 15);

    // per-thread staging addresses
    const int rA0 = tid >> 2,         sA0 = (tid & 3) * 8;
    const int rA1 = (tid + 256) >> 2, sA1 = (tid & 3) * 8;
    const int rB  = tid >> 2,         sB  = (tid & 3) * 8;
    const unsigned lA0[2] = { (unsigned)(uintptr_t)&As[0][rA0 * 32 + sA0],
                              (unsigned)(uintptr_t)&As[1][rA0 * 32 + sA0] };
    const unsigned lA1[2] = { (unsigned)(uintptr_t)&As[0][rA1 * 32 + sA1],
                              (unsigned)(uintptr_t)&As[1][rA1 * 32 + sA1] };
    const unsigned lB[2]  = { (unsigned)(uintptr_t)&Bs[0][rB * 32 + sB],
                              (unsigned)(uintptr_t)&Bs[1][rB * 32 + sB] };
    const bf16* gA0 = &X[(size_t)(Mbase + rA0) * E_ + sA0];
    const bf16* gA1 = &X[(size_t)(Mbase + rA1) * E_ + sA1];
    const bf16* gB  = &W[(size_t)(Nbase + rB)  * E_ + sB];

    // prologue: stage tile 0
    async_g2l_b128(lA0[0], gA0);
    async_g2l_b128(lA1[0], gA1);
    async_g2l_b128(lB[0],  gB);

    for (int kt = 0; kt < E_ / 32; ++kt) {
        const int cur = kt & 1;
        if (kt + 1 < E_ / 32) {
            const int nxt = (kt + 1) & 1, kn = (kt + 1) * 32;
            async_g2l_b128(lA0[nxt], gA0 + kn);
            async_g2l_b128(lA1[nxt], gA1 + kn);
            async_g2l_b128(lB[nxt],  gB + kn);
            if (kt + 2 < E_ / 32) {  // warm L2 one slice ahead of the DMA
                __builtin_prefetch(gA0 + kn + 32, 0, 1);
                __builtin_prefetch(gB + kn + 32, 0, 1);
            }
            wait_async3();           // tile kt landed; tile kt+1 still in flight
        } else {
            wait_async0();
        }
        __syncthreads();

        const bf16* Ac = As[cur];
        const bf16* Bc = Bs[cur];
        v16bf a0 = cat8(*(const v8bf*)&Ac[arow * 32 + koffA],
                        *(const v8bf*)&Ac[arow * 32 + koffA + 8]);
        v16bf a1 = cat8(*(const v8bf*)&Ac[(arow + 16) * 32 + koffA],
                        *(const v8bf*)&Ac[(arow + 16) * 32 + koffA + 8]);
        v16bf b0 = *(const v16bf*)&Bc[bcol * 32 + koffB];
        v16bf b1 = *(const v16bf*)&Bc[(bcol + 16) * 32 + koffB];

        acc00 = __builtin_amdgcn_wmma_f32_16x16x32_bf16(false, a0, false, b0, (short)0, acc00, false, false);
        acc01 = __builtin_amdgcn_wmma_f32_16x16x32_bf16(false, a0, false, b1, (short)0, acc01, false, false);
        acc10 = __builtin_amdgcn_wmma_f32_16x16x32_bf16(false, a1, false, b0, (short)0, acc10, false, false);
        acc11 = __builtin_amdgcn_wmma_f32_16x16x32_bf16(false, a1, false, b1, (short)0, acc11, false, false);
        __syncthreads();   // buffer cur free for reuse at iteration kt+2
    }

    const int rbase = (lane < 16) ? 0 : 8;     // C/D layout: lanes 0-15 M=r, 16-31 M=8+r
    const int ncol  = lane & 15;
    #pragma unroll
    for (int r = 0; r < 8; ++r) {
        int m = Mbase + m0w + rbase + r;
        int n = Nbase + n0w + ncol;
        XP[(size_t)m * G_ + n]             = acc00[r] + bias[n];
        XP[(size_t)m * G_ + n + 16]        = acc01[r] + bias[n + 16];
        XP[(size_t)(m + 16) * G_ + n]      = acc10[r] + bias[n];
        XP[(size_t)(m + 16) * G_ + n + 16] = acc11[r] + bias[n + 16];
    }
}

// --------------------- persistent bidirectional LSTM scan (one launch total)
// grid = (16, 2): blockIdx.y = direction, each block owns hidden chunk [j0,j0+32)
// for all T steps; cell state lives in registers; per-step cross-WG ordering via
// per-step release/acquire counters (+ HW cluster barrier when clustered).
__global__ void __launch_bounds__(256)
lstm_scan(const bf16* __restrict__ WhhF, const bf16* __restrict__ WhhB,
          const float* __restrict__ XPf, const float* __restrict__ XPb,
          float* __restrict__ hseq,
          bf16* __restrict__ hF0, bf16* __restrict__ hF1,
          bf16* __restrict__ hB0, bf16* __restrict__ hB1,
          unsigned* __restrict__ barBase) {
    const int dir = blockIdx.y;
    const bf16*  Whh  = dir ? WhhB : WhhF;
    const float* XP   = dir ? XPb  : XPf;
    bf16* buf0        = dir ? hB0  : hF0;   // zero-initialized (h0 = 0)
    bf16* buf1        = dir ? hB1  : hF1;
    unsigned* bar     = barBase + dir * T_;
    const int dirOff  = dir ? H_ : 0;

    __shared__ float gb[4][32][33];
    const int tid = threadIdx.x, lane = tid & 31, wv = tid >> 5;
    const int j0 = blockIdx.x * 32;
    const int gi = wv >> 1, half = wv & 1;
    const int bcol  = gi * H_ + j0 + half * 16 + (lane & 15);
    const int arow  = lane & 15;
    const int koffA = (lane >> 4) * 16;
    const int koffB = (lane >> 4) * 16;

    float creg[4] = {0.f, 0.f, 0.f, 0.f};   // cell state for this thread's 4 (b,j)

    for (int t = 0; t < T_; ++t) {
        const int tt = dir ? (T_ - 1 - t) : t;
        const bf16* hp = (t & 1) ? buf1 : buf0;
        bf16*       hw = (t & 1) ? buf0 : buf1;

        v8f acc0 = {}, acc1 = {};
        for (int kt = 0; kt < H_ / 32; ++kt) {
            const int k0 = kt * 32;
            v16bf a0 = cat8(*(const v8bf*)&hp[(size_t)arow * H_ + k0 + koffA],
                            *(const v8bf*)&hp[(size_t)arow * H_ + k0 + koffA + 8]);
            v16bf a1 = cat8(*(const v8bf*)&hp[(size_t)(arow + 16) * H_ + k0 + koffA],
                            *(const v8bf*)&hp[(size_t)(arow + 16) * H_ + k0 + koffA + 8]);
            v16bf bb = *(const v16bf*)&Whh[(size_t)bcol * H_ + k0 + koffB];
            acc0 = __builtin_amdgcn_wmma_f32_16x16x32_bf16(false, a0, false, bb, (short)0, acc0, false, false);
            acc1 = __builtin_amdgcn_wmma_f32_16x16x32_bf16(false, a1, false, bb, (short)0, acc1, false, false);
        }

        const int jj    = half * 16 + (lane & 15);
        const int gcol  = gi * H_ + j0 + jj;
        const int rbase = (lane < 16) ? 0 : 8;
        #pragma unroll
        for (int r = 0; r < 8; ++r) {
            int b0 = rbase + r;
            gb[gi][b0][jj]      = acc0[r] + XP[((size_t)b0 * T_ + tt) * G_ + gcol];
            gb[gi][b0 + 16][jj] = acc1[r] + XP[((size_t)(b0 + 16) * T_ + tt) * G_ + gcol];
        }
        __syncthreads();

        #pragma unroll
        for (int q = 0; q < 4; ++q) {
            int idx = tid + q * 256;
            int b = idx >> 5, jj2 = idx & 31;
            int j = j0 + jj2;
            float iv = gb[0][b][jj2], fv = gb[1][b][jj2];
            float gv = gb[2][b][jj2], ov = gb[3][b][jj2];
            float cn = sigmf(fv) * creg[q] + sigmf(iv) * tanhf(gv);
            float hv = sigmf(ov) * tanhf(cn);
            creg[q] = cn;
            hseq[((size_t)b * T_ + tt) * (2 * H_) + dirOff + j] = hv;
            hw[b * H_ + j] = (bf16)hv;
        }

        // make this block's h-slice visible, then wait for all 16 blocks of this dir
        __threadfence();
        __syncthreads();
        if (tid == 0) {
#if __has_builtin(__builtin_amdgcn_s_cluster_barrier)
            __builtin_amdgcn_s_cluster_barrier();  // HW path for 16-WG cluster dispatch (NOP otherwise)
#endif
            __hip_atomic_fetch_add(&bar[t], 1u, __ATOMIC_RELEASE, __HIP_MEMORY_SCOPE_AGENT);
            while (__hip_atomic_load(&bar[t], __ATOMIC_ACQUIRE, __HIP_MEMORY_SCOPE_AGENT) < (unsigned)NBLK_SCAN)
                __builtin_amdgcn_s_sleep(1);
        }
        __syncthreads();
    }
}

// --------------------------------------------- classifier logits + CRF -llh/B
__global__ void __launch_bounds__(288)
cls_crf(const float* __restrict__ hseq, const float* __restrict__ clsW,
        const float* __restrict__ clsB, const int* __restrict__ label,
        const float* __restrict__ startT, const float* __restrict__ endT,
        const float* __restrict__ trans, float* __restrict__ out) {
    __shared__ float logit[C_];
    const int b = blockIdx.x;
    const int tid = threadIdx.x, lane = tid & 31, wv = tid >> 5;
    float num = 0.f;
    float alpha[C_];

    for (int t = 0; t < T_; ++t) {
        if (wv < C_) {
            const float* h = &hseq[((size_t)b * T_ + t) * (2 * H_)];
            const float* w = &clsW[(size_t)wv * (2 * H_)];
            float p = 0.f;
            for (int q = lane; q < 2 * H_; q += 32) p += h[q] * w[q];
            #pragma unroll
            for (int off = 16; off > 0; off >>= 1) p += __shfl_down(p, off, 32);
            if (lane == 0) logit[wv] = p + clsB[wv];
        }
        __syncthreads();
        if (tid == 0) {
            int tg = label[b * T_ + t];
            if (t == 0) {
                int t0 = (tg > -1) ? tg : 0;
                num = startT[t0] + logit[t0];
                #pragma unroll
                for (int c = 0; c < C_; ++c) alpha[c] = startT[c] + logit[c];
            } else {
                int tgp = label[b * T_ + t - 1];
                if (tg > -1 && tgp > -1)
                    num += logit[tg] + trans[tgp * C_ + tg];
                float nxt[C_];
                #pragma unroll
                for (int c2 = 0; c2 < C_; ++c2) {
                    float mx = -1e30f;
                    #pragma unroll
                    for (int c = 0; c < C_; ++c)
                        mx = fmaxf(mx, alpha[c] + trans[c * C_ + c2]);
                    float s = 0.f;
                    #pragma unroll
                    for (int c = 0; c < C_; ++c)
                        s += __expf(alpha[c] + trans[c * C_ + c2] - mx);
                    nxt[c2] = mx + __logf(s) + logit[c2];
                }
                if (tg > -1)
                    #pragma unroll
                    for (int c2 = 0; c2 < C_; ++c2) alpha[c2] = nxt[c2];
            }
        }
        __syncthreads();
    }

    if (tid == 0) {
        int cnt = 0;
        for (int t = 0; t < T_; ++t) if (label[b * T_ + t] > -1) cnt++;
        int li = (cnt > 0) ? cnt - 1 : 0;
        int lastTag = label[b * T_ + li];
        if (lastTag < 0) lastTag = 0;
        num += endT[lastTag];
        float mx = -1e30f;
        #pragma unroll
        for (int c = 0; c < C_; ++c) mx = fmaxf(mx, alpha[c] + endT[c]);
        float s = 0.f;
        #pragma unroll
        for (int c = 0; c < C_; ++c) s += __expf(alpha[c] + endT[c] - mx);
        float den = mx + __logf(s);
        atomicAdd(out, -(num - den) / (float)B_);
    }
}

// --------------------------------------------------------------------- launch
extern "C" void kernel_launch(void* const* d_in, const int* in_sizes, int n_in,
                              void* d_out, int out_size, void* d_ws, size_t ws_size,
                              hipStream_t stream) {
    (void)in_sizes; (void)n_in; (void)out_size; (void)ws_size;
    const int*   input  = (const int*)  d_in[0];
    const int*   label  = (const int*)  d_in[1];
    const float* emb    = (const float*)d_in[2];
    const float* w_ih_f = (const float*)d_in[3];
    const float* w_hh_f = (const float*)d_in[4];
    const float* b_ih_f = (const float*)d_in[5];
    const float* b_hh_f = (const float*)d_in[6];
    const float* w_ih_b = (const float*)d_in[7];
    const float* w_hh_b = (const float*)d_in[8];
    const float* b_ih_b = (const float*)d_in[9];
    const float* b_hh_b = (const float*)d_in[10];
    const float* cls_w  = (const float*)d_in[11];
    const float* cls_b  = (const float*)d_in[12];
    const float* startT = (const float*)d_in[13];
    const float* endT   = (const float*)d_in[14];
    const float* trans  = (const float*)d_in[15];
    float* out = (float*)d_out;

    char* ws = (char*)d_ws;
    size_t off = 0;
    auto take = [&](size_t bytes) -> char* {
        char* p = ws + off;
        off = (off + bytes + 255) & ~(size_t)255;
        return p;
    };
    bf16*  xbf   = (bf16*) take((size_t)M_ * E_ * 2);
    bf16*  wihF  = (bf16*) take((size_t)G_ * E_ * 2);
    bf16*  whhF  = (bf16*) take((size_t)G_ * H_ * 2);
    bf16*  wihB  = (bf16*) take((size_t)G_ * E_ * 2);
    bf16*  whhB  = (bf16*) take((size_t)G_ * H_ * 2);
    float* biasF = (float*)take((size_t)G_ * 4);
    float* biasB = (float*)take((size_t)G_ * 4);
    float* XPf   = (float*)take((size_t)M_ * G_ * 4);
    float* XPb   = (float*)take((size_t)M_ * G_ * 4);
    float* hseq  = (float*)take((size_t)M_ * 2 * H_ * 4);
    bf16*  hF0   = (bf16*) take((size_t)B_ * H_ * 2);
    bf16*  hF1   = (bf16*) take((size_t)B_ * H_ * 2);
    bf16*  hB0   = (bf16*) take((size_t)B_ * H_ * 2);
    bf16*  hB1   = (bf16*) take((size_t)B_ * H_ * 2);
    unsigned* bar = (unsigned*)take((size_t)2 * T_ * 4);

    zero_init<<<(B_ * H_ + 255) / 256, 256, 0, stream>>>(hF0, hB0, bar, out, B_ * H_);
    embed_gather<<<M_, 256, 0, stream>>>(input, emb, xbf);
    cvt_bf16<<<(G_ * E_ + 255) / 256, 256, 0, stream>>>(w_ih_f, wihF, G_ * E_);
    cvt_bf16<<<(G_ * H_ + 255) / 256, 256, 0, stream>>>(w_hh_f, whhF, G_ * H_);
    cvt_bf16<<<(G_ * E_ + 255) / 256, 256, 0, stream>>>(w_ih_b, wihB, G_ * E_);
    cvt_bf16<<<(G_ * H_ + 255) / 256, 256, 0, stream>>>(w_hh_b, whhB, G_ * H_);
    bias_sum<<<(G_ + 255) / 256, 256, 0, stream>>>(b_ih_f, b_hh_f, biasF, G_);
    bias_sum<<<(G_ + 255) / 256, 256, 0, stream>>>(b_ih_b, b_hh_b, biasB, G_);

    dim3 ggrid(G_ / 64, M_ / 128, 2);
    xproj_gemm<<<ggrid, 256, 0, stream>>>(xbf, wihF, wihB, biasF, biasB, XPf, XPb);

    lstm_scan<<<dim3(NBLK_SCAN, 2), 256, 0, stream>>>(whhF, whhB, XPf, XPb, hseq,
                                                      hF0, hF1, hB0, hB1, bar);

    cls_crf<<<B_, 288, 0, stream>>>(hseq, cls_w, cls_b, label, startT, endT, trans, out);
}
